// MoELayer_46291157516846
// MI455X (gfx1250) — compile-verified, matrix operand order checked by reading, not run on hardware
//
#include <hip/hip_runtime.h>
#include <hip/hip_bf16.h>

// ---------------------------------------------------------------------------
// MoE layer (E=8, C=768, F=3072, top-2) for gfx1250 (MI455X).
// Sparse routed grouped-GEMM, fused FFN (GEMM1 + exact-erf GELU + GEMM2).
// Weights pre-swizzled to bf16 WMMA B-fragment layout (2x b128 per fragment).
// TM=32 tokens/block, 16 wave32: each B fragment feeds 2 WMMAs (both M
// halves) -> 1 VMEM instr per WMMA, halved L2 weight traffic.
// ---------------------------------------------------------------------------

typedef __attribute__((ext_vector_type(16))) __bf16 v16bf;
typedef __attribute__((ext_vector_type(8)))  __bf16 v8bf;
typedef __attribute__((ext_vector_type(8)))  float  v8f;

#define NEXP   8
#define CDIM   768
#define FDIM   3072
#define TM     32            // token rows per block tile
#define KSTEP  32            // K per bf16 WMMA
#define FCHUNK 768           // F processed per chunk (H tile kept in LDS)
#define NCHUNK (FDIM / FCHUNK)
#define LDX    (CDIM + 8)    // LDS row stride (bf16 elems), keeps 16B align
#define LDH    (FCHUNK + 8)
#define KT1    (CDIM / 32)   // 24  K-tiles in GEMM1
#define KT2    (FDIM / 32)   // 96  K-tiles in GEMM2
#define NWAVE  16
#define NTHREADS (NWAVE * 32)

// ---------------------------------------------------------------------------
// Zero output + expert counters
// ---------------------------------------------------------------------------
__global__ void moe_zero_kernel(float* __restrict__ out, size_t n,
                                int* __restrict__ cnt) {
    size_t i = (size_t)blockIdx.x * blockDim.x + threadIdx.x;
    size_t stride = (size_t)gridDim.x * blockDim.x;
    for (; i < n; i += stride) out[i] = 0.0f;
    if (blockIdx.x == 0 && threadIdx.x < NEXP) cnt[threadIdx.x] = 0;
}

// ---------------------------------------------------------------------------
// Pre-swizzle fp32 weights [E][K][Nn] (row-major) into bf16 WMMA B-fragment
// layout: 32x16 (KxN) tiles, kt-contiguous per N-tile; within a tile the
// 512 bf16 are stored lane-major (lane*16 + elem) so a wave fetches one
// fragment as a contiguous 32B per lane (two global_load_b128).
//   lane l (l<16):  col n = l,     K = {0..7, 16..23}
//   lane l (l>=16): col n = l-16,  K = {8..15, 24..31}
// ---------------------------------------------------------------------------
__global__ void moe_swizzle_kernel(const float* __restrict__ W,
                                   __bf16* __restrict__ out,
                                   int K, int Nn, size_t total_chunks) {
    size_t g = (size_t)blockIdx.x * blockDim.x + threadIdx.x;
    size_t stride = (size_t)gridDim.x * blockDim.x;
    const int tiles_k = K >> 5;          // K/32
    const int tiles_n = Nn >> 4;         // Nn/16
    const int tiles_per_e = tiles_k * tiles_n;
    for (; g < total_chunks; g += stride) {
        size_t tile = g >> 5;            // global tile id
        int lane = (int)(g & 31);
        int e   = (int)(tile / tiles_per_e);
        int rem = (int)(tile % tiles_per_e);
        int nt  = rem / tiles_k;         // N-tile (kt contiguous inside)
        int kt  = rem % tiles_k;
        const float* We = W + (size_t)e * K * Nn;
        __bf16* op = out + tile * 512 + (size_t)lane * 16;
        int n = nt * 16 + (lane & 15);
        int kbase = kt * 32 + ((lane >> 4) << 3);   // +8 for hi half-wave
#pragma unroll
        for (int ee = 0; ee < 16; ++ee) {
            int k = kbase + (ee < 8 ? ee : ee + 8); // {0..7,16..23} rel.
            op[ee] = (__bf16)We[(size_t)k * Nn + n];
        }
    }
}

// ---------------------------------------------------------------------------
// Router: logits = x @ router_w^T, top-2, renormalized softmax weights.
// ---------------------------------------------------------------------------
__global__ __launch_bounds__(256)
void moe_router_kernel(const float* __restrict__ x,
                       const float* __restrict__ rw,
                       int* __restrict__ topi, float* __restrict__ topw,
                       int N) {
    int t = blockIdx.x * blockDim.x + threadIdx.x;
    if (t >= N) return;
    float acc[NEXP];
#pragma unroll
    for (int e = 0; e < NEXP; ++e) acc[e] = 0.0f;
    const float* xr = x + (size_t)t * CDIM;
    for (int c = 0; c < CDIM; ++c) {
        float xv = xr[c];
#pragma unroll
        for (int e = 0; e < NEXP; ++e) acc[e] += xv * rw[e * CDIM + c];
    }
    int i1 = 0;                                     // ties -> lower index
#pragma unroll
    for (int e = 1; e < NEXP; ++e) if (acc[e] > acc[i1]) i1 = e;
    int i2 = (i1 == 0) ? 1 : 0;
#pragma unroll
    for (int e = 0; e < NEXP; ++e)
        if (e != i1 && acc[e] > acc[i2]) i2 = e;
    float p2 = __expf(acc[i2] - acc[i1]);
    float inv = __builtin_amdgcn_rcpf(1.0f + p2);
    topi[t * 2 + 0] = i1;  topw[t * 2 + 0] = inv;
    topi[t * 2 + 1] = i2;  topw[t * 2 + 1] = p2 * inv;
}

// ---------------------------------------------------------------------------
// Scatter tokens into per-expert buckets.
// ---------------------------------------------------------------------------
__global__ __launch_bounds__(256)
void moe_scatter_kernel(const int* __restrict__ topi,
                        const float* __restrict__ topw,
                        int* __restrict__ cnt,
                        int* __restrict__ toklist, float* __restrict__ wlist,
                        int N) {
    int t = blockIdx.x * blockDim.x + threadIdx.x;
    if (t >= N) return;
#pragma unroll
    for (int k = 0; k < 2; ++k) {
        int e = topi[t * 2 + k];
        int pos = atomicAdd(&cnt[e], 1);
        toklist[(size_t)e * N + pos] = t;
        wlist[(size_t)e * N + pos] = topw[t * 2 + k];
    }
}

// ---------------------------------------------------------------------------
// Branchless exact-erf GELU (Abramowitz-Stegun 7.1.26, |err| < 1.5e-7).
// ---------------------------------------------------------------------------
__device__ __forceinline__ float gelu_erf(float v) {
    float s = v * 0.70710678118654752f;
    float a = __builtin_fabsf(s);
    float t = __builtin_amdgcn_rcpf(__builtin_fmaf(0.3275911f, a, 1.0f));
    float p = __builtin_fmaf(t, 1.061405429f, -1.453152027f);
    p = __builtin_fmaf(t, p, 1.421413741f);
    p = __builtin_fmaf(t, p, -0.284496736f);
    p = __builtin_fmaf(t, p, 0.254829592f);
    p = p * t;
    float ex = __expf(-a * a);
    float erfabs = __builtin_fmaf(-p, ex, 1.0f);
    float erfv = __builtin_copysignf(erfabs, s);
    return 0.5f * v * (1.0f + erfv);
}

__device__ __forceinline__ v16bf cat8(v8bf lo, v8bf hi) {
    return __builtin_shufflevector(lo, hi, 0, 1, 2, 3, 4, 5, 6, 7,
                                           8, 9, 10, 11, 12, 13, 14, 15);
}

// ---------------------------------------------------------------------------
// Fused expert FFN on a 32-token tile. 512 threads = 16 wave32.
// Each wave: 3 N-tiles x 2 M-halves; every B fragment feeds 2 WMMAs.
//   H = gelu(Xg @ W1[e] + b1[e])  (F chunked; H chunk in LDS as bf16)
//   out[tok] += w_tok * (H @ W2[e] + b2[e])
// ---------------------------------------------------------------------------
__global__ __launch_bounds__(NTHREADS)
void moe_ffn_kernel(const float* __restrict__ x,
                    const __bf16* __restrict__ w1s,   // swizzled bf16
                    const float* __restrict__ b1,
                    const __bf16* __restrict__ w2s,   // swizzled bf16
                    const float* __restrict__ b2,
                    const int* __restrict__ cnt,
                    const int* __restrict__ toklist,
                    const float* __restrict__ wlist,
                    float* __restrict__ out,
                    int N, int tiles_per_e) {
    __shared__ __bf16 Xs[TM * LDX];    // 49.7 KB
    __shared__ __bf16 Hs[TM * LDH];    // 49.7 KB
    __shared__ int   s_tok[TM];
    __shared__ float s_wt[TM];

    const int e    = blockIdx.x / tiles_per_e;
    const int tile = blockIdx.x % tiles_per_e;
    const int ne   = cnt[e];
    if (tile * TM >= ne) return;

    const int tid    = threadIdx.x;
    const int lane   = tid & 31;
    const int wv     = tid >> 5;          // wave 0..15
    const int halfHi = (lane >> 4) & 1;
    const int lan16  = lane & 15;

    if (tid < TM) {
        int idx = tile * TM + tid;
        if (idx < ne) {
            s_tok[tid] = toklist[(size_t)e * N + idx];
            s_wt[tid]  = wlist[(size_t)e * N + idx];
        } else {
            s_tok[tid] = 0;
            s_wt[tid]  = 0.0f;
        }
    }
    __syncthreads();

    // Gather X rows (fp32 -> bf16) into LDS, float4-vectorized.
    for (int i = tid; i < TM * (CDIM / 4); i += NTHREADS) {
        int r = i / (CDIM / 4), c4 = (i - r * (CDIM / 4)) * 4;
        float4 v = *(const float4*)&x[(size_t)s_tok[r] * CDIM + c4];
        __bf16* dst = &Xs[r * LDX + c4];
        dst[0] = (__bf16)v.x; dst[1] = (__bf16)v.y;
        dst[2] = (__bf16)v.z; dst[3] = (__bf16)v.w;
    }
    __syncthreads();

    const __bf16* W1e = w1s + (size_t)e * CDIM * FDIM;   // swizzled
    const __bf16* W2e = w2s + (size_t)e * FDIM * CDIM;   // swizzled
    const float* b1e = b1 + (size_t)e * FDIM;
    const float* b2e = b2 + (size_t)e * CDIM;
    const size_t lane_off = (size_t)lane * 16;

    v8f oacc[2][3];
#pragma unroll
    for (int h = 0; h < 2; ++h)
#pragma unroll
        for (int j = 0; j < 3; ++j) oacc[h][j] = (v8f){};

    for (int fc = 0; fc < NCHUNK; ++fc) {
        const int f0 = fc * FCHUNK;

        // ---- Phase A: H chunk = gelu(Xs @ W1[:, f0:f0+768] + b1) ----------
        v8f hacc[2][3];
#pragma unroll
        for (int h = 0; h < 2; ++h)
#pragma unroll
            for (int j = 0; j < 3; ++j) hacc[h][j] = (v8f){};

#pragma unroll 1
        for (int k0 = 0; k0 < CDIM; k0 += KSTEP) {
            const int kt = k0 >> 5;
            const __bf16* ap0 = &Xs[lan16 * LDX + k0 + halfHi * 8];
            const __bf16* ap1 = ap0 + 16 * LDX;
            v16bf a0 = cat8(*(const v8bf*)ap0, *(const v8bf*)(ap0 + 16));
            v16bf a1 = cat8(*(const v8bf*)ap1, *(const v8bf*)(ap1 + 16));
            __builtin_prefetch(
                W1e + ((size_t)(fc * 48 + wv) * KT1 + kt + 4) * 512, 0, 1);
#pragma unroll
            for (int jj = 0; jj < 3; ++jj) {
                const int ntile = fc * 48 + wv + jj * NWAVE; // global F N-tile
                v16bf b = *(const v16bf*)(
                    W1e + ((size_t)ntile * KT1 + kt) * 512 + lane_off);
                hacc[0][jj] = __builtin_amdgcn_wmma_f32_16x16x32_bf16(
                                  false, a0, false, b, (short)0, hacc[0][jj],
                                  false, false);
                hacc[1][jj] = __builtin_amdgcn_wmma_f32_16x16x32_bf16(
                                  false, a1, false, b, (short)0, hacc[1][jj],
                                  false, false);
            }
        }
        // epilogue: bias + GELU -> Hs (bf16)
#pragma unroll
        for (int jj = 0; jj < 3; ++jj) {
            const int j = wv + jj * NWAVE;
            const int n = f0 + j * 16 + lan16;
            const float bias = b1e[n];
            const int nl = j * 16 + lan16;
#pragma unroll
            for (int h = 0; h < 2; ++h) {
#pragma unroll
                for (int r = 0; r < 8; ++r) {
                    int m = h * 16 + r + halfHi * 8;
                    Hs[m * LDH + nl] = (__bf16)gelu_erf(hacc[h][jj][r] + bias);
                }
            }
        }
        __syncthreads();

        // ---- Phase B: oacc += Hs @ W2[f0:f0+768, :] -----------------------
#pragma unroll 1
        for (int k0 = 0; k0 < FCHUNK; k0 += KSTEP) {
            const int kt = (f0 + k0) >> 5;
            const __bf16* ap0 = &Hs[lan16 * LDH + k0 + halfHi * 8];
            const __bf16* ap1 = ap0 + 16 * LDH;
            v16bf a0 = cat8(*(const v8bf*)ap0, *(const v8bf*)(ap0 + 16));
            v16bf a1 = cat8(*(const v8bf*)ap1, *(const v8bf*)(ap1 + 16));
            __builtin_prefetch(
                W2e + ((size_t)wv * KT2 + kt + 4) * 512, 0, 1);
#pragma unroll
            for (int jj = 0; jj < 3; ++jj) {
                const int ntile = wv + jj * NWAVE;           // out C N-tile
                v16bf b = *(const v16bf*)(
                    W2e + ((size_t)ntile * KT2 + kt) * 512 + lane_off);
                oacc[0][jj] = __builtin_amdgcn_wmma_f32_16x16x32_bf16(
                                  false, a0, false, b, (short)0, oacc[0][jj],
                                  false, false);
                oacc[1][jj] = __builtin_amdgcn_wmma_f32_16x16x32_bf16(
                                  false, a1, false, b, (short)0, oacc[1][jj],
                                  false, false);
            }
        }
        __syncthreads();
    }

    // ---- Combine: out[tok] += w_tok * (oacc + b2) --------------------------
#pragma unroll
    for (int jj = 0; jj < 3; ++jj) {
        const int n = (wv + jj * NWAVE) * 16 + lan16;
        const float bias = b2e[n];
#pragma unroll
        for (int h = 0; h < 2; ++h) {
#pragma unroll
            for (int r = 0; r < 8; ++r) {
                int m = h * 16 + r + halfHi * 8;
                float v = (oacc[h][jj][r] + bias) * s_wt[m];
                atomicAdd(&out[(size_t)s_tok[m] * CDIM + n], v);
            }
        }
    }
}

// ---------------------------------------------------------------------------
// Launch
// ---------------------------------------------------------------------------
extern "C" void kernel_launch(void* const* d_in, const int* in_sizes, int n_in,
                              void* d_out, int out_size, void* d_ws, size_t ws_size,
                              hipStream_t stream) {
    const float* x   = (const float*)d_in[0];
    const float* rw  = (const float*)d_in[1];
    const float* w1  = (const float*)d_in[2];
    const float* b1  = (const float*)d_in[3];
    const float* w2  = (const float*)d_in[4];
    const float* b2  = (const float*)d_in[5];
    float* out = (float*)d_out;

    const int N = in_sizes[0] / CDIM;   // 16384 tokens

    // Workspace layout: cnt | topi | topw | toklist | wlist | w1s | w2s
    char* ws = (char*)d_ws;
    size_t off = 0;
    int* cnt = (int*)(ws + off);                 off += 256;
    int* topi = (int*)(ws + off);                off += (size_t)N * 2 * 4;
    float* topw = (float*)(ws + off);            off += (size_t)N * 2 * 4;
    int* toklist = (int*)(ws + off);             off += (size_t)NEXP * N * 4;
    float* wlist = (float*)(ws + off);           off += (size_t)NEXP * N * 4;
    off = (off + 255) & ~(size_t)255;
    __bf16* w1s = (__bf16*)(ws + off);           off += (size_t)NEXP * CDIM * FDIM * 2;
    __bf16* w2s = (__bf16*)(ws + off);           off += (size_t)NEXP * FDIM * CDIM * 2;

    const size_t outn = (size_t)N * CDIM;
    moe_zero_kernel<<<256, 256, 0, stream>>>(out, outn, cnt);

    const size_t chunks1 = (size_t)NEXP * CDIM * FDIM / 16;  // (tile,lane) units
    moe_swizzle_kernel<<<2048, 256, 0, stream>>>(w1, w1s, CDIM, FDIM, chunks1);
    moe_swizzle_kernel<<<2048, 256, 0, stream>>>(w2, w2s, FDIM, CDIM, chunks1);

    moe_router_kernel<<<(N + 255) / 256, 256, 0, stream>>>(x, rw, topi, topw, N);
    moe_scatter_kernel<<<(N + 255) / 256, 256, 0, stream>>>(topi, topw, cnt,
                                                            toklist, wlist, N);
    const int tiles = (N + TM - 1) / TM;
    moe_ffn_kernel<<<NEXP * tiles, NTHREADS, 0, stream>>>(x, w1s, b1, w2s, b2,
                                                          cnt, toklist, wlist,
                                                          out, N, tiles);
}
